// RCNNHead_22497038696824
// MI455X (gfx1250) — matrix operand order repaired and checked by reading, not compiled, not run on hardware
//
#include <hip/hip_runtime.h>

// ---------------- problem constants (fixed by reference setup) ----------------
#define OUTD   7
#define NCLS   21
#define HIDN   1024
#define CCH    512
#define FH     50
#define FW     50
#define NIMG   2
#define SROI   512
#define RTOT   (NIMG*SROI)        // 1024 rows
#define DFEAT  (CCH*OUTD*OUTD)    // 25088
#define DETN   100
#define CAND   (SROI*(NCLS-1))    // 10240 candidates per image
#define LOG_MAXV 3.91202300542814598f  // log(50)

typedef __attribute__((ext_vector_type(16))) __bf16 v16bf;
typedef __attribute__((ext_vector_type(8)))  float  v8f;

union ABfrag { uint4 q[2]; v16bf v; };   // 32 bytes: 16 bf16 fragment

__device__ __forceinline__ unsigned short f2bf(float f) {
    unsigned int u = __float_as_uint(f);
    u += 0x7fffu + ((u >> 16) & 1u);     // round-to-nearest-even
    return (unsigned short)(u >> 16);
}

// ---------------- 1) weight transpose + fp32 -> bf16 pack --------------------
// in:  w  [K][Nn] fp32 (row-major) ; out: wt [Nn][K] bf16
__global__ void pack_wt_kernel(const float* __restrict__ w,
                               unsigned short* __restrict__ wt,
                               int K, int Nn) {
    int idx = blockIdx.x * blockDim.x + threadIdx.x;
    if (idx >= K * Nn) return;
    int n = idx / K;
    int k = idx - n * K;
    wt[(size_t)n * K + k] = f2bf(w[(size_t)k * Nn + n]);
}

// ---------------- 2) ROI-align -> X bf16 [RTOT][DFEAT] -----------------------
// one block per (roi, 7x7 cell); threads sweep channels
__global__ void roi_pack_kernel(const float* __restrict__ feat,
                                const float* __restrict__ props,
                                unsigned short* __restrict__ X) {
    int blk  = blockIdx.x;           // r*49 + cell
    int r    = blk / 49;
    int cell = blk - r * 49;
    int py   = cell / OUTD;
    int px   = cell - py * OUTD;
    int img  = r / SROI;

    const float* p = props + (size_t)r * 4;
    float x1 = p[0] * 0.0625f, y1 = p[1] * 0.0625f;
    float x2 = p[2] * 0.0625f, y2 = p[3] * 0.0625f;
    float bw = fmaxf(x2 - x1, 1.0f) * (1.0f / (float)OUTD);
    float bh = fmaxf(y2 - y1, 1.0f) * (1.0f / (float)OUTD);

    int   iy0[2], iy1[2], ix0[2], ix1[2];
    float ly[2], lx[2];
    bool  vy[2], vx[2];
#pragma unroll
    for (int i = 0; i < 2; ++i) {
        float gy = (float)py + ((float)i + 0.5f) * 0.5f;
        float y  = y1 + gy * bh;
        vy[i] = (y > -1.0f) && (y < (float)FH);
        float yc = fminf(fmaxf(y, 0.0f), (float)(FH - 1));
        int y0 = (int)floorf(yc);
        iy0[i] = y0; iy1[i] = min(y0 + 1, FH - 1); ly[i] = yc - (float)y0;

        float gx = (float)px + ((float)i + 0.5f) * 0.5f;
        float x  = x1 + gx * bw;
        vx[i] = (x > -1.0f) && (x < (float)FW);
        float xc = fminf(fmaxf(x, 0.0f), (float)(FW - 1));
        int x0 = (int)floorf(xc);
        ix0[i] = x0; ix1[i] = min(x0 + 1, FW - 1); lx[i] = xc - (float)x0;
    }

    const float* fb = feat + (size_t)img * CCH * FH * FW;
    for (int c = threadIdx.x; c < CCH; c += blockDim.x) {
        const float* fc = fb + (size_t)c * FH * FW;
        float acc = 0.0f;
#pragma unroll
        for (int i = 0; i < 2; ++i)
#pragma unroll
            for (int j = 0; j < 2; ++j) {
                float f00 = fc[iy0[i] * FW + ix0[j]];
                float f01 = fc[iy0[i] * FW + ix1[j]];
                float f10 = fc[iy1[i] * FW + ix0[j]];
                float f11 = fc[iy1[i] * FW + ix1[j]];
                float v = f00 * (1.0f - ly[i]) * (1.0f - lx[j])
                        + f01 * (1.0f - ly[i]) * lx[j]
                        + f10 * ly[i] * (1.0f - lx[j])
                        + f11 * ly[i] * lx[j];
                acc += (vy[i] && vx[j]) ? v : 0.0f;
            }
        X[(size_t)r * DFEAT + c * 49 + cell] = f2bf(acc * 0.25f);
    }
}

// ---------------- 3) bf16 WMMA GEMM:  out[M][Nn] = A[M][K] @ Bt[Nn][K]^T ------
// 8 waves/block; wave grid 4(M) x 2(N); each wave: 2x2 tiles of 16x16.
// Workgroup tile: 128(M) x 64(N). Requires M%128==0, Nn%64==0, K%32==0.
__global__ __launch_bounds__(256) void gemm_bf16_wmma(
    const unsigned short* __restrict__ A,
    const unsigned short* __restrict__ Bt,
    const float* __restrict__ bias,
    float* __restrict__ outf,
    unsigned short* __restrict__ outb,
    int M, int Nn, int K, int relu) {
    const int tid  = threadIdx.x;
    const int lane = tid & 31;
    const int wave = tid >> 5;
    const int l15  = lane & 15;
    const int kh   = lane >> 4;          // lane half selects K sub-range
    const int wm   = wave >> 1;          // 0..3
    const int wn   = wave & 1;           // 0..1
    const int m0   = blockIdx.y * 128 + wm * 32;
    const int n0   = blockIdx.x * 64  + wn * 32;

    const v8f z = {0.f,0.f,0.f,0.f,0.f,0.f,0.f,0.f};
    v8f acc[2][2];
    acc[0][0] = z; acc[0][1] = z; acc[1][0] = z; acc[1][1] = z;

    // A 16-bit fragment per lane: K = 16*(e>>3) + 8*half + (e&7)
    const unsigned short* a0p = A  + (size_t)(m0 + l15) * K + 8 * kh;
    const unsigned short* a1p = a0p + (size_t)16 * K;
    // B 16-bit fragment per lane: K = 16*(e>>3) + 8*half + (e&7)
    const unsigned short* b0p = Bt + (size_t)(n0 + l15) * K + 8 * kh;
    const unsigned short* b1p = b0p + (size_t)16 * K;

    for (int k = 0; k < K; k += 32) {
        ABfrag a0, a1, b0, b1;
        a0.q[0] = *(const uint4*)(a0p + k);      // K: 8h .. 8h+7
        a0.q[1] = *(const uint4*)(a0p + k + 16); // K: 16+8h .. 16+8h+7
        a1.q[0] = *(const uint4*)(a1p + k);
        a1.q[1] = *(const uint4*)(a1p + k + 16);
        b0.q[0] = *(const uint4*)(b0p + k);      // K: 8h .. 8h+7
        b0.q[1] = *(const uint4*)(b0p + k + 16); // K: 16+8h .. 16+8h+7
        b1.q[0] = *(const uint4*)(b1p + k);
        b1.q[1] = *(const uint4*)(b1p + k + 16);

        acc[0][0] = __builtin_amdgcn_wmma_f32_16x16x32_bf16(
            false, a0.v, false, b0.v, (short)0, acc[0][0], false, false);
        acc[0][1] = __builtin_amdgcn_wmma_f32_16x16x32_bf16(
            false, a0.v, false, b1.v, (short)0, acc[0][1], false, false);
        acc[1][0] = __builtin_amdgcn_wmma_f32_16x16x32_bf16(
            false, a1.v, false, b0.v, (short)0, acc[1][0], false, false);
        acc[1][1] = __builtin_amdgcn_wmma_f32_16x16x32_bf16(
            false, a1.v, false, b1.v, (short)0, acc[1][1], false, false);
    }

    // C/D layout: lane l -> N = l&15 ; VGPR r -> M = r + 8*(l>>4)
#pragma unroll
    for (int mt = 0; mt < 2; ++mt) {
#pragma unroll
        for (int nt = 0; nt < 2; ++nt) {
            int nc = n0 + nt * 16 + l15;
            float bv = bias ? bias[nc] : 0.0f;
#pragma unroll
            for (int r = 0; r < 8; ++r) {
                int mr = m0 + mt * 16 + r + 8 * kh;
                float v = acc[mt][nt][r] + bv;
                if (relu) v = fmaxf(v, 0.0f);
                size_t o = (size_t)mr * Nn + nc;
                if (outf) outf[o] = v;
                if (outb) outb[o] = f2bf(v);
            }
        }
    }
}

// ---------------- 4) heads: cls/box dots + softmax + decode + clip -----------
__global__ __launch_bounds__(128) void heads_kernel(
    const float* __restrict__ H2,
    const float* __restrict__ wc, const float* __restrict__ bc,
    const float* __restrict__ wb, const float* __restrict__ bb,
    const float* __restrict__ props, const int* __restrict__ img_sz,
    float* __restrict__ scores, float* __restrict__ boxdec) {
    __shared__ float hrow[HIDN];
    __shared__ float logit[NCLS];
    __shared__ float delta[NCLS * 4];
    __shared__ float red[2];

    int r = blockIdx.x;
    const float* h = H2 + (size_t)r * HIDN;
    for (int k = threadIdx.x; k < HIDN; k += blockDim.x) hrow[k] = h[k];
    __syncthreads();

    int j = threadIdx.x;
    if (j < NCLS) {
        float s = bc[j];
        for (int k = 0; k < HIDN; ++k) s += hrow[k] * wc[k * NCLS + j];
        logit[j] = s;
    } else if (j < NCLS + NCLS * 4) {
        int j2 = j - NCLS;
        float s = bb[j2];
        for (int k = 0; k < HIDN; ++k) s += hrow[k] * wb[k * (NCLS * 4) + j2];
        delta[j2] = s;
    }
    __syncthreads();

    if (threadIdx.x == 0) {
        float m = logit[0];
        for (int k = 1; k < NCLS; ++k) m = fmaxf(m, logit[k]);
        float ssum = 0.0f;
        for (int k = 0; k < NCLS; ++k) ssum += expf(logit[k] - m);
        red[0] = m; red[1] = ssum;
    }
    __syncthreads();

    if (j < NCLS) {
        scores[(size_t)r * NCLS + j] = expf(logit[j] - red[0]) / red[1];

        const float* p = props + (size_t)r * 4;
        float pw  = p[2] - p[0], ph = p[3] - p[1];
        float pcx = p[0] + 0.5f * pw, pcy = p[1] + 0.5f * ph;
        float dx = delta[j * 4 + 0] * 0.1f;
        float dy = delta[j * 4 + 1] * 0.1f;
        float dw = fminf(delta[j * 4 + 2] * 0.2f, LOG_MAXV);
        float dh = fminf(delta[j * 4 + 3] * 0.2f, LOG_MAXV);
        float cx = dx * pw + pcx, cy = dy * ph + pcy;
        float w  = expf(dw) * pw, hh = expf(dh) * ph;
        float img = (float)(*img_sz);
        float* bd = boxdec + ((size_t)r * NCLS + j) * 4;
        bd[0] = fminf(fmaxf(cx - 0.5f * w,  0.0f), img);
        bd[1] = fminf(fmaxf(cy - 0.5f * hh, 0.0f), img);
        bd[2] = fminf(fmaxf(cx + 0.5f * w,  0.0f), img);
        bd[3] = fminf(fmaxf(cy + 0.5f * hh, 0.0f), img);
    }
}

// ---------------- 5) NMS: one block per image --------------------------------
__global__ __launch_bounds__(256) void nms_kernel(
    const float* __restrict__ scores, const float* __restrict__ boxdec,
    const int* __restrict__ img_sz,
    float* __restrict__ live, float* __restrict__ nbox,
    float* __restrict__ areas, float* __restrict__ out) {
    __shared__ float bestv[256];
    __shared__ int   besti[256];
    __shared__ float sb[5];   // chosen nb box (4) + score

    int img_i = blockIdx.x;
    float img = (float)(*img_sz);
    float* lv = live  + (size_t)img_i * CAND;
    float* nb = nbox  + (size_t)img_i * CAND * 4;
    float* ar = areas + (size_t)img_i * CAND;

    for (int j = threadIdx.x; j < CAND; j += 256) {
        int s   = j / 20;
        int cls = 1 + (j - s * 20);
        int r   = img_i * SROI + s;
        float sc = scores[(size_t)r * NCLS + cls];
        const float* b = boxdec + ((size_t)r * NCLS + cls) * 4;
        float wv = b[2] - b[0], hv = b[3] - b[1];
        bool valid = (sc > 0.05f) && (wv >= 0.01f) && (hv >= 0.01f);
        lv[j] = valid ? sc : -1.0f;
        float off = (float)cls * (img + 1.0f);
        float n0 = b[0] + off, n1 = b[1] + off, n2 = b[2] + off, n3 = b[3] + off;
        nb[j * 4 + 0] = n0; nb[j * 4 + 1] = n1;
        nb[j * 4 + 2] = n2; nb[j * 4 + 3] = n3;
        ar[j] = (n2 - n0) * (n3 - n1);
    }
    __syncthreads();

    float* out_boxes  = out;                       // [NIMG][DETN][4]
    float* out_scores = out + NIMG * DETN * 4;     // [NIMG][DETN]
    int*   out_labels = (int*)(out + NIMG * DETN * 4 + NIMG * DETN);

    for (int step = 0; step < DETN; ++step) {
        // argmax with first-index tie-break (matches jnp.argmax)
        float bv = -1e38f; int bi = 0x7fffffff;
        for (int j = threadIdx.x; j < CAND; j += 256) {
            float v = lv[j];
            if (v > bv) { bv = v; bi = j; }
        }
        bestv[threadIdx.x] = bv; besti[threadIdx.x] = bi;
        __syncthreads();
        for (int sft = 128; sft > 0; sft >>= 1) {
            if (threadIdx.x < sft) {
                float v2 = bestv[threadIdx.x + sft];
                int   i2 = besti[threadIdx.x + sft];
                if (v2 > bestv[threadIdx.x] ||
                    (v2 == bestv[threadIdx.x] && i2 < besti[threadIdx.x])) {
                    bestv[threadIdx.x] = v2; besti[threadIdx.x] = i2;
                }
            }
            __syncthreads();
        }
        if (threadIdx.x == 0) {
            int   i = besti[0];
            float s = bestv[0];
            sb[0] = nb[i * 4 + 0]; sb[1] = nb[i * 4 + 1];
            sb[2] = nb[i * 4 + 2]; sb[3] = nb[i * 4 + 3];
            sb[4] = s;
            int o = img_i * DETN + step;
            bool keep = s > 0.05f;
            int sC = i / 20, cls = 1 + (i - sC * 20);
            int r = img_i * SROI + sC;
            const float* b = boxdec + ((size_t)r * NCLS + cls) * 4;
            out_boxes[o * 4 + 0] = keep ? b[0] : 0.0f;
            out_boxes[o * 4 + 1] = keep ? b[1] : 0.0f;
            out_boxes[o * 4 + 2] = keep ? b[2] : 0.0f;
            out_boxes[o * 4 + 3] = keep ? b[3] : 0.0f;
            out_scores[o] = keep ? s : 0.0f;
            out_labels[o] = keep ? cls : 0;
        }
        __syncthreads();
        float c0 = sb[0], c1 = sb[1], c2 = sb[2], c3 = sb[3];
        float area_i = (c2 - c0) * (c3 - c1);
        for (int j = threadIdx.x; j < CAND; j += 256) {
            float xx1 = fmaxf(c0, nb[j * 4 + 0]);
            float yy1 = fmaxf(c1, nb[j * 4 + 1]);
            float xx2 = fminf(c2, nb[j * 4 + 2]);
            float yy2 = fminf(c3, nb[j * 4 + 3]);
            float inter = fmaxf(xx2 - xx1, 0.0f) * fmaxf(yy2 - yy1, 0.0f);
            float iou = inter / (area_i + ar[j] - inter + 1e-9f);
            if (iou > 0.5f) lv[j] = -1.0f;
        }
        __syncthreads();
    }
}

// ---------------- host-side launch -------------------------------------------
extern "C" void kernel_launch(void* const* d_in, const int* in_sizes, int n_in,
                              void* d_out, int out_size, void* d_ws, size_t ws_size,
                              hipStream_t stream) {
    const float* features  = (const float*)d_in[0];
    const float* proposals = (const float*)d_in[1];
    const float* w1 = (const float*)d_in[2];
    const float* b1 = (const float*)d_in[3];
    const float* w2 = (const float*)d_in[4];
    const float* b2 = (const float*)d_in[5];
    const float* wc = (const float*)d_in[6];
    const float* bc = (const float*)d_in[7];
    const float* wb = (const float*)d_in[8];
    const float* bb = (const float*)d_in[9];
    const int*   imgsz = (const int*)d_in[10];

    char* ws = (char*)d_ws;
    size_t o = 0;
    unsigned short* X   = (unsigned short*)(ws + o); o += (size_t)RTOT * DFEAT * 2;  // 51.4 MB
    unsigned short* W1T = (unsigned short*)(ws + o); o += (size_t)HIDN * DFEAT * 2;  // 51.4 MB
    unsigned short* W2T = (unsigned short*)(ws + o); o += (size_t)HIDN * HIDN * 2;   // 2 MB
    unsigned short* H1  = (unsigned short*)(ws + o); o += (size_t)RTOT * HIDN * 2;   // 2 MB
    float* H2   = (float*)(ws + o); o += (size_t)RTOT * HIDN * 4;                    // 4 MB
    float* SC   = (float*)(ws + o); o += (size_t)RTOT * NCLS * 4;
    float* BD   = (float*)(ws + o); o += (size_t)RTOT * NCLS * 4 * 4;
    float* LIVE = (float*)(ws + o); o += (size_t)NIMG * CAND * 4;
    float* NB   = (float*)(ws + o); o += (size_t)NIMG * CAND * 16;
    float* AR   = (float*)(ws + o); o += (size_t)NIMG * CAND * 4;

    // weight packs (independent of data; deterministic each call)
    pack_wt_kernel<<<(DFEAT * HIDN + 255) / 256, 256, 0, stream>>>(w1, W1T, DFEAT, HIDN);
    pack_wt_kernel<<<(HIDN * HIDN + 255) / 256, 256, 0, stream>>>(w2, W2T, HIDN, HIDN);

    // ROI-align directly into bf16 GEMM operand layout
    roi_pack_kernel<<<RTOT * 49, 256, 0, stream>>>(features, proposals, X);

    // FC1: X[1024][25088] @ W1T -> H1 (bf16, fused bias+ReLU)
    gemm_bf16_wmma<<<dim3(HIDN / 64, RTOT / 128), 256, 0, stream>>>(
        X, W1T, b1, nullptr, H1, RTOT, HIDN, DFEAT, 1);
    // FC2: H1 @ W2T -> H2 (fp32, fused bias+ReLU)
    gemm_bf16_wmma<<<dim3(HIDN / 64, RTOT / 128), 256, 0, stream>>>(
        H1, W2T, b2, H2, nullptr, RTOT, HIDN, HIDN, 1);

    // cls/box heads + softmax + decode + clip
    heads_kernel<<<RTOT, 128, 0, stream>>>(H2, wc, bc, wb, bb, proposals, imgsz, SC, BD);

    // per-image NMS -> d_out (boxes | scores | labels)
    nms_kernel<<<NIMG, 256, 0, stream>>>(SC, BD, imgsz, LIVE, NB, AR, (float*)d_out);
}
